// GNN_dot_product_67035849556080
// MI455X (gfx1250) — compile-verified, matrix operand order checked by reading, not run on hardware
//
#include <hip/hip_runtime.h>
#include <math.h>

typedef __attribute__((ext_vector_type(16))) __bf16 v16bf;
typedef __attribute__((ext_vector_type(8)))  __bf16 v8bf;
typedef __attribute__((ext_vector_type(8)))  float  v8f;
typedef __attribute__((ext_vector_type(4)))  int    v4i;

typedef __attribute__((address_space(1))) v4i gv4i;   // global
typedef __attribute__((address_space(3))) v4i lv4i;   // LDS

#define K_IN   512     // IN_CH
#define HC     512     // HEADS*CH
#define NHEADS 8
#define NCOL   1024    // concatenated xl|xr columns
#define BPANEL 32768   // packed-B elements (bf16) per 64-column panel

#if __has_builtin(__builtin_amdgcn_global_load_async_to_lds_b128)
#define HAVE_ASYNC_LDS 1
#else
#define HAVE_ASYNC_LDS 0
#endif

// ---------------------------------------------------------------------------
__device__ __forceinline__ unsigned short f2bf(float f) {
    unsigned u = __float_as_uint(f);
    unsigned r = (u + 0x7FFFu + ((u >> 16) & 1u)) >> 16;
    return (unsigned short)r;
}

// monotone-int-punned atomic float max (mixed signs OK, init = -inf)
__device__ __forceinline__ void atomicMaxFloat(float* addr, float val) {
    if (val >= 0.0f) atomicMax((int*)addr, __float_as_int(val));
    else             atomicMin((unsigned int*)addr, __float_as_uint(val));
}

// ---------------------------------------------------------------------------
// Pack W_l|W_r into bf16 B-fragment layout, panel-contiguous:
//   pw[ n64 * 32768 + (k32*4 + j) * 512 + L*16 + e ]
// lane L owns column n = n64*64 + j*16 + (L%16); e=0..7 -> K = kb+(L/16)*8+e,
// e=8..15 -> K = kb+16+(L/16)*8+(e-8). 32 contiguous bytes per lane.
__global__ void pack_w_kernel(const float* __restrict__ Wl,
                              const float* __restrict__ Wr,
                              unsigned short* __restrict__ pw) {
    int idx = blockIdx.x * 256 + threadIdx.x;
    if (idx >= K_IN * NCOL) return;
    int n64  = idx >> 15;
    int tile = (idx >> 9) & 63;       // k32*4 + j
    int r    = idx & 511;
    int L = r >> 4, e = r & 15;
    int k32 = tile >> 2, j = tile & 3;
    int h = L >> 4;
    int n = n64 * 64 + j * 16 + (L & 15);
    int k = k32 * 32 + h * 8 + (e & 7) + ((e >> 3) << 4);
    float v = (n < 512) ? Wl[k * 512 + n] : Wr[k * 512 + (n - 512)];
    pw[idx] = f2bf(v);
}

// ---------------------------------------------------------------------------
__global__ void init_kernel(float* __restrict__ agg, float* __restrict__ denom,
                            float* __restrict__ amax, float* __restrict__ gsum,
                            float* __restrict__ gsq, long long nagg, int n8) {
    long long i = (long long)blockIdx.x * 256 + threadIdx.x;
    if (i < nagg) agg[i] = 0.0f;
    if (i < n8) { denom[i] = 0.0f; amax[i] = -__builtin_inff(); }
    if (i < HC) { gsum[i] = 0.0f; gsq[i] = 0.0f; }
}

// ---------------------------------------------------------------------------
// WMMA GEMM: xlxr[N,1024] = bf16(x) @ bf16(W_l|W_r), f32 accumulate.
// Block = 8 waves = 128 rows x 64 cols. The 64-col B panel (64 KB) is staged
// in LDS once per block via gfx1250 async global->LDS copies; each wave then
// computes a 16x64 strip with 4 accumulators, A double-buffered from global.
__global__ void gemm_wmma_kernel(const float* __restrict__ x,
                                 const unsigned short* __restrict__ pw,
                                 float* __restrict__ xlxr, int N) {
    extern __shared__ unsigned short ldsB[];   // 32768 bf16 = 64 KB

    int tid  = threadIdx.x;
    int lane = tid & 31;
    int wv   = tid >> 5;
    int n64  = blockIdx.x & 15;
    int mg   = blockIdx.x >> 4;

    // ---- stage B panel: 64 KB = 4096 x b128, 16 chunks per thread ----
    const unsigned short* gsrc = pw + (size_t)n64 * BPANEL;
#if HAVE_ASYNC_LDS
#pragma unroll
    for (int i = 0; i < 16; ++i) {
        int off = (i * 256 + tid) * 8;   // element offset, 16B chunks
        __builtin_amdgcn_global_load_async_to_lds_b128(
            (gv4i*)(gsrc + off), (lv4i*)(ldsB + off), 0, 0);
    }
#if __has_builtin(__builtin_amdgcn_s_wait_asynccnt)
    __builtin_amdgcn_s_wait_asynccnt(0);
#else
    asm volatile("s_wait_asynccnt 0x0" ::: "memory");
#endif
#else
#pragma unroll
    for (int i = 0; i < 16; ++i) {
        int off = (i * 256 + tid) * 8;
        *(float4*)(ldsB + off) = *(const float4*)(gsrc + off);
    }
#endif
    __syncthreads();

    // ---- per-wave 16x64 strip ----
    int mbase = mg * 128 + wv * 16;
    int l = lane & 15, h = lane >> 4;
    int rowA = mbase + l; if (rowA >= N) rowA = N - 1;
    const float* arow = x + (size_t)rowA * K_IN;

    v8f acc0 = {}, acc1 = {}, acc2 = {}, acc3 = {};

    // prime A double buffer (four 16B chunks per lane per k-block)
    const float4* p0 = (const float4*)(arow + h * 8);
    const float4* p1 = (const float4*)(arow + 16 + h * 8);
    float4 c0 = p0[0], c1 = p0[1], c2 = p1[0], c3 = p1[1];

    for (int kb = 0; kb < K_IN; kb += 32) {
        float4 n0, n1, n2, n3;
        if (kb + 32 < K_IN) {
            const float4* q0 = (const float4*)(arow + kb + 32 + h * 8);
            const float4* q1 = (const float4*)(arow + kb + 48 + h * 8);
            n0 = q0[0]; n1 = q0[1]; n2 = q1[0]; n3 = q1[1];
        }
        float fa[16] = { c0.x, c0.y, c0.z, c0.w, c1.x, c1.y, c1.z, c1.w,
                         c2.x, c2.y, c2.z, c2.w, c3.x, c3.y, c3.z, c3.w };
        v16bf a;
#pragma unroll
        for (int i = 0; i < 16; ++i) a[i] = (__bf16)fa[i];

        const unsigned short* lb = ldsB + ((kb >> 5) * 4) * 512 + lane * 16;
        v16bf b0 = __builtin_shufflevector(*(const v8bf*)(lb +    0), *(const v8bf*)(lb +    8),
                                           0,1,2,3,4,5,6,7,8,9,10,11,12,13,14,15);
        v16bf b1 = __builtin_shufflevector(*(const v8bf*)(lb +  512), *(const v8bf*)(lb +  520),
                                           0,1,2,3,4,5,6,7,8,9,10,11,12,13,14,15);
        v16bf b2 = __builtin_shufflevector(*(const v8bf*)(lb + 1024), *(const v8bf*)(lb + 1032),
                                           0,1,2,3,4,5,6,7,8,9,10,11,12,13,14,15);
        v16bf b3 = __builtin_shufflevector(*(const v8bf*)(lb + 1536), *(const v8bf*)(lb + 1544),
                                           0,1,2,3,4,5,6,7,8,9,10,11,12,13,14,15);

        acc0 = __builtin_amdgcn_wmma_f32_16x16x32_bf16(false, a, false, b0, (short)0, acc0, false, false);
        acc1 = __builtin_amdgcn_wmma_f32_16x16x32_bf16(false, a, false, b1, (short)0, acc1, false, false);
        acc2 = __builtin_amdgcn_wmma_f32_16x16x32_bf16(false, a, false, b2, (short)0, acc2, false, false);
        acc3 = __builtin_amdgcn_wmma_f32_16x16x32_bf16(false, a, false, b3, (short)0, acc3, false, false);

        c0 = n0; c1 = n1; c2 = n2; c3 = n3;
    }

    // ---- store: VGPR r -> M = r (lanes 0-15) / r+8 (lanes 16-31) ----
    int col  = n64 * 64 + l;
    int rtop = mbase + h * 8;
    if (mbase + 15 < N) {
#pragma unroll
        for (int r = 0; r < 8; ++r) {
            size_t o = (size_t)(rtop + r) * NCOL + col;
            xlxr[o +  0] = acc0[r];
            xlxr[o + 16] = acc1[r];
            xlxr[o + 32] = acc2[r];
            xlxr[o + 48] = acc3[r];
        }
    } else {
#pragma unroll
        for (int r = 0; r < 8; ++r) {
            int row = rtop + r;
            if (row < N) {
                size_t o = (size_t)row * NCOL + col;
                xlxr[o +  0] = acc0[r];
                xlxr[o + 16] = acc1[r];
                xlxr[o + 32] = acc2[r];
                xlxr[o + 48] = acc3[r];
            }
        }
    }
}

// ---------------------------------------------------------------------------
// Per-edge attention logits: e = leaky_relu(xl[src]+xr[dst]); alpha = e . att
__global__ void attn_kernel(const float* __restrict__ xlxr, const int* __restrict__ ei,
                            const float* __restrict__ att, float* __restrict__ logits,
                            float* __restrict__ amax, int E) {
    int lane = threadIdx.x & 31;
    int e = blockIdx.x * (blockDim.x >> 5) + (threadIdx.x >> 5);
    if (e >= E) return;
    int src = ei[e], dst = ei[E + e];
    const float4* pl = (const float4*)(xlxr + (size_t)src * NCOL + lane * 16);
    const float4* pr = (const float4*)(xlxr + (size_t)dst * NCOL + 512 + lane * 16);
    const float4* pa = (const float4*)(att + lane * 16);
    float s = 0.0f;
#pragma unroll
    for (int q = 0; q < 4; ++q) {
        float4 a = pl[q], b = pr[q], w = pa[q];
        float v;
        v = a.x + b.x; v = v > 0.0f ? v : 0.2f * v; s += v * w.x;
        v = a.y + b.y; v = v > 0.0f ? v : 0.2f * v; s += v * w.y;
        v = a.z + b.z; v = v > 0.0f ? v : 0.2f * v; s += v * w.z;
        v = a.w + b.w; v = v > 0.0f ? v : 0.2f * v; s += v * w.w;
    }
    s += __shfl_xor(s, 1, 32);
    s += __shfl_xor(s, 2, 32);
    if ((lane & 3) == 0) {
        int hd = lane >> 2;
        logits[(size_t)e * NHEADS + hd] = s;
        atomicMaxFloat(&amax[(size_t)dst * NHEADS + hd], s);
    }
}

__global__ void expsum_kernel(const int* __restrict__ ei, float* __restrict__ logits,
                              const float* __restrict__ amax, float* __restrict__ denom,
                              int E) {
    int t = blockIdx.x * 256 + threadIdx.x;
    if (t >= E * NHEADS) return;
    int e = t >> 3, hd = t & 7;
    int dst = ei[E + e];
    float v = __expf(logits[t] - amax[(size_t)dst * NHEADS + hd]);
    logits[t] = v;
    atomicAdd(&denom[(size_t)dst * NHEADS + hd], v);
}

__global__ void aggregate_kernel(const float* __restrict__ xlxr, const int* __restrict__ ei,
                                 const float* __restrict__ logits, const float* __restrict__ denom,
                                 float* __restrict__ agg, int E) {
    int lane = threadIdx.x & 31;
    int e = blockIdx.x * (blockDim.x >> 5) + (threadIdx.x >> 5);
    if (e >= E) return;
    int src = ei[e], dst = ei[E + e];
    int hd = lane >> 2;
    float w = logits[(size_t)e * NHEADS + hd] /
              (denom[(size_t)dst * NHEADS + hd] + 1e-16f);
    const float4* ps = (const float4*)(xlxr + (size_t)src * NCOL + lane * 16);
    float* po = agg + (size_t)dst * HC + lane * 16;
#pragma unroll
    for (int q = 0; q < 4; ++q) {
        float4 v = ps[q];
        atomicAdd(po + q * 4 + 0, w * v.x);
        atomicAdd(po + q * 4 + 1, w * v.y);
        atomicAdd(po + q * 4 + 2, w * v.z);
        atomicAdd(po + q * 4 + 3, w * v.w);
    }
}

__global__ void bnpartial_kernel(const float* __restrict__ agg, float* __restrict__ gsum,
                                 float* __restrict__ gsq, int N) {
    int tid = threadIdx.x;
    int rows = (N + gridDim.x - 1) / gridDim.x;
    int r0 = blockIdx.x * rows;
    int r1 = r0 + rows; if (r1 > N) r1 = N;
    float s0 = 0, q0 = 0, s1 = 0, q1 = 0;
    for (int r = r0; r < r1; ++r) {
        float v0 = agg[(size_t)r * HC + tid];
        float v1 = agg[(size_t)r * HC + tid + 256];
        s0 += v0; q0 += v0 * v0;
        s1 += v1; q1 += v1 * v1;
    }
    atomicAdd(&gsum[tid], s0);       atomicAdd(&gsq[tid], q0);
    atomicAdd(&gsum[tid + 256], s1); atomicAdd(&gsq[tid + 256], q1);
}

__global__ void bnfinal_kernel(const float* __restrict__ gsum, const float* __restrict__ gsq,
                               const float* __restrict__ gamma, const float* __restrict__ beta,
                               float* __restrict__ chanA, float* __restrict__ chanB, int N) {
    int c = blockIdx.x * 256 + threadIdx.x;
    if (c >= HC) return;
    float inv  = 1.0f / (float)N;
    float mean = gsum[c] * inv;
    float var  = gsq[c] * inv - mean * mean;
    float rstd = rsqrtf(var + 1e-5f);
    float a = gamma[c] * rstd;
    chanA[c] = a;
    chanB[c] = beta[c] - mean * a;
}

__global__ void normrelu_kernel(float* __restrict__ agg, const float* __restrict__ chanA,
                                const float* __restrict__ chanB, long long total) {
    long long i = (long long)blockIdx.x * 256 + threadIdx.x;
    if (i >= total) return;
    int c = (int)(i & (HC - 1));
    float y = agg[i] * chanA[c] + chanB[c];
    agg[i] = y > 0.0f ? y : 0.0f;
}

__global__ void score_kernel(const float* __restrict__ y, const int* __restrict__ eli,
                             float* __restrict__ out, int EL) {
    int lane = threadIdx.x & 31;
    int e = blockIdx.x * (blockDim.x >> 5) + (threadIdx.x >> 5);
    if (e >= EL) return;
    int s = eli[e], t = eli[EL + e];
    const float4* ps = (const float4*)(y + (size_t)s * HC + lane * 16);
    const float4* pt = (const float4*)(y + (size_t)t * HC + lane * 16);
    float d = 0.0f;
#pragma unroll
    for (int q = 0; q < 4; ++q) {
        float4 a = ps[q], b = pt[q];
        d += a.x * b.x + a.y * b.y + a.z * b.z + a.w * b.w;
    }
    d += __shfl_xor(d, 1, 32);
    d += __shfl_xor(d, 2, 32);
    d += __shfl_xor(d, 4, 32);
    d += __shfl_xor(d, 8, 32);
    d += __shfl_xor(d, 16, 32);
    if (lane == 0) out[e] = d;
}

// ---------------------------------------------------------------------------
extern "C" void kernel_launch(void* const* d_in, const int* in_sizes, int n_in,
                              void* d_out, int out_size, void* d_ws, size_t ws_size,
                              hipStream_t stream) {
    const float* x     = (const float*)d_in[0];
    const int*   ei    = (const int*)d_in[1];
    const int*   eli   = (const int*)d_in[2];
    const float* Wl    = (const float*)d_in[3];
    const float* Wr    = (const float*)d_in[4];
    const float* att   = (const float*)d_in[5];
    const float* gamma = (const float*)d_in[7];
    const float* beta  = (const float*)d_in[8];
    float* out = (float*)d_out;

    int N  = in_sizes[0] / K_IN;
    int E  = in_sizes[1] / 2;
    int EL = in_sizes[2] / 2;

    char* w = (char*)d_ws;
    unsigned short* pw = (unsigned short*)w;                    // 512*1024 bf16 = 1 MiB
    float* xlxr  = (float*)(w + (size_t)K_IN * NCOL * 2);       // N*1024 f32
    float* logits = xlxr + (size_t)N * NCOL;                    // E*8
    float* amax   = logits + (size_t)E * NHEADS;                // N*8
    float* denom  = amax + (size_t)N * NHEADS;                  // N*8
    float* agg    = denom + (size_t)N * NHEADS;                 // N*512
    float* gsum   = agg + (size_t)N * HC;                       // 512
    float* gsq    = gsum + HC;                                  // 512
    float* chanA  = gsq + HC;                                   // 512
    float* chanB  = chanA + HC;                                 // 512

    long long nagg = (long long)N * HC;

    pack_w_kernel<<<(K_IN * NCOL + 255) / 256, 256, 0, stream>>>(Wl, Wr, pw);

    init_kernel<<<(unsigned)((nagg + 255) / 256), 256, 0, stream>>>(
        agg, denom, amax, gsum, gsq, nagg, N * NHEADS);

    int mgroups = (N + 127) / 128;
    gemm_wmma_kernel<<<mgroups * 16, 256, 65536, stream>>>(x, pw, xlxr, N);

    attn_kernel<<<(E + 7) / 8, 256, 0, stream>>>(xlxr, ei, att, logits, amax, E);

    expsum_kernel<<<(E * NHEADS + 255) / 256, 256, 0, stream>>>(ei, logits, amax, denom, E);

    aggregate_kernel<<<(E + 7) / 8, 256, 0, stream>>>(xlxr, ei, logits, denom, agg, E);

    bnpartial_kernel<<<200, 256, 0, stream>>>(agg, gsum, gsq, N);
    bnfinal_kernel<<<2, 256, 0, stream>>>(gsum, gsq, gamma, beta, chanA, chanB, N);

    normrelu_kernel<<<(unsigned)((nagg + 255) / 256), 256, 0, stream>>>(agg, chanA, chanB, nagg);

    score_kernel<<<(EL + 7) / 8, 256, 0, stream>>>(agg, eli, out, EL);
}
